// TopKSAE_53618371723773
// MI455X (gfx1250) — compile-verified, matrix operand order checked by reading, not run on hardware
//
#include <hip/hip_runtime.h>

typedef unsigned short ushort_t;
typedef unsigned int uint_t;

typedef __attribute__((ext_vector_type(16))) __bf16 v16bf;
typedef __attribute__((ext_vector_type(8)))  float  v8f;

#define N_TOK   16384
#define D_MODEL 1024
#define D_DICT  16384
#define K_TOP   32

// ---------------- fp32 -> bf16 conversion (round-to-nearest-even) ------------
__device__ __forceinline__ ushort_t f2bf(float f) {
  uint_t u = __float_as_uint(f);
  u += 0x7FFFu + ((u >> 16) & 1u);
  return (ushort_t)(u >> 16);
}

__global__ void cvt_bf16_kernel(const float* __restrict__ src,
                                ushort_t* __restrict__ dst, long long n) {
  long long i = ((long long)blockIdx.x * blockDim.x + threadIdx.x) * 4;
  if (i + 3 < n) {
    float4 f = *(const float4*)(src + i);
    ushort_t o0 = f2bf(f.x), o1 = f2bf(f.y), o2 = f2bf(f.z), o3 = f2bf(f.w);
    uint_t lo = (uint_t)o0 | ((uint_t)o1 << 16);
    uint_t hi = (uint_t)o2 | ((uint_t)o3 << 16);
    *(uint2*)(dst + i) = make_uint2(lo, hi);
  }
}

// ---------------- W_dec [D_MODEL, D_DICT] -> WdT [D_DICT, D_MODEL] -----------
__global__ void transpose_wdec_kernel(const float* __restrict__ Wd,
                                      float* __restrict__ WdT) {
  __shared__ float tile[32][33];
  int f0 = blockIdx.x * 32;           // along D_DICT
  int d0 = blockIdx.y * 32;           // along D_MODEL
  int tx = threadIdx.x & 31, ty = threadIdx.x >> 5;   // 256 thr = 32x8
  for (int r = ty; r < 32; r += 8)
    tile[r][tx] = Wd[(size_t)(d0 + r) * D_DICT + f0 + tx];
  __syncthreads();
  for (int r = ty; r < 32; r += 8)
    WdT[(size_t)(f0 + r) * D_MODEL + d0 + tx] = tile[tx][r];
}

// ---------------- Encode GEMM: z = x @ W_enc^T + b_enc (bf16 WMMA) ----------
#define BM 128
#define BN 256
#define KC 64
#define LDA 72   // padded LDS stride (elems): 144B rows, 16B aligned
#define LDB 72

#define A_CHUNKS (BM * KC / 8 / 256)   // 4 x 16B chunks per thread
#define B_CHUNKS (BN * KC / 8 / 256)   // 8 x 16B chunks per thread

union FragU { uint4 q[2]; v16bf v; };

__launch_bounds__(256)
__global__ void sae_encode_wmma(const ushort_t* __restrict__ Xb,
                                const ushort_t* __restrict__ Wb,
                                const float* __restrict__ b_enc,
                                float* __restrict__ Z) {
  __shared__ ushort_t As[BM * LDA];
  __shared__ ushort_t Bs[BN * LDB];

  const int tid  = threadIdx.x;
  const int wid  = tid >> 5;
  const int lane = tid & 31;
  const int wm = wid & 1;           // 2 wave-rows  (64 rows each)
  const int wn = wid >> 1;          // 4 wave-cols  (64 cols each)
  const int laneRow  = lane & 15;
  const int laneHalf = lane >> 4;

  const int row0 = blockIdx.y * BM;
  const int col0 = blockIdx.x * BN;

  v8f acc[4][4] = {};

  for (int k0 = 0; k0 < D_MODEL; k0 += KC) {
    // global -> LDS, current K-chunk
    for (int i = 0; i < A_CHUNKS; ++i) {
      int c = tid + i * 256, r = c >> 3, c8 = (c & 7) << 3;
      uint4 v = *(const uint4*)(Xb + (size_t)(row0 + r) * D_MODEL + k0 + c8);
      *(uint4*)(As + r * LDA + c8) = v;
    }
    for (int i = 0; i < B_CHUNKS; ++i) {
      int c = tid + i * 256, r = c >> 3, c8 = (c & 7) << 3;
      uint4 v = *(const uint4*)(Wb + (size_t)(col0 + r) * D_MODEL + k0 + c8);
      *(uint4*)(Bs + r * LDB + c8) = v;
    }

    // prefetch next K-chunk into near cache (global_prefetch_b8: no VGPR,
    // no counter cost) so next iteration's loads hit L0/WGP$ under the WMMAs
    int kn = k0 + KC;
    if (kn < D_MODEL) {
      for (int i = 0; i < A_CHUNKS; ++i) {
        int c = tid + i * 256, r = c >> 3, c8 = (c & 7) << 3;
        __builtin_prefetch(Xb + (size_t)(row0 + r) * D_MODEL + kn + c8, 0, 3);
      }
      for (int i = 0; i < B_CHUNKS; ++i) {
        int c = tid + i * 256, r = c >> 3, c8 = (c & 7) << 3;
        __builtin_prefetch(Wb + (size_t)(col0 + r) * D_MODEL + kn + c8, 0, 3);
      }
    }
    __syncthreads();

    for (int kk = 0; kk < KC; kk += 32) {
      FragU a[4], b[4];
      for (int i = 0; i < 4; ++i) {
        const ushort_t* pA = As + (wm * 64 + i * 16 + laneRow) * LDA + kk + laneHalf * 8;
        a[i].q[0] = *(const uint4*)pA;
        a[i].q[1] = *(const uint4*)(pA + 16);
        const ushort_t* pB = Bs + (wn * 64 + i * 16 + laneRow) * LDB + kk + laneHalf * 8;
        b[i].q[0] = *(const uint4*)pB;
        b[i].q[1] = *(const uint4*)(pB + 16);
      }
      for (int mi = 0; mi < 4; ++mi)
        for (int ni = 0; ni < 4; ++ni)
          acc[mi][ni] = __builtin_amdgcn_wmma_f32_16x16x32_bf16(
              false, a[mi].v, false, b[ni].v, (short)0, acc[mi][ni],
              false, false);
    }
    __syncthreads();
  }

  // epilogue: + b_enc, store z
  for (int mi = 0; mi < 4; ++mi) {
    for (int ni = 0; ni < 4; ++ni) {
      int nc = col0 + wn * 64 + ni * 16 + laneRow;
      float bb = b_enc[nc];
      for (int v = 0; v < 8; ++v) {
        int m = row0 + wm * 64 + mi * 16 + v + 8 * laneHalf;
        Z[(size_t)m * D_DICT + nc] = acc[mi][ni][v] + bb;
      }
    }
  }
}

// ---------------- Top-k (k=32) per row, in-place sparsify --------------------
#define TK_NT 512

__global__ void sae_topk_kernel(float* __restrict__ Z,
                                int* __restrict__ outIdx,
                                float* __restrict__ outVal) {
  extern __shared__ unsigned char tk_smem[];
  float* rowb = (float*)tk_smem;            // D_DICT floats
  float* redV = rowb + D_DICT;              // TK_NT
  int*   redI = (int*)(redV + TK_NT);       // TK_NT
  float* lVal = (float*)(redI + TK_NT);     // K_TOP
  int*   lIdx = (int*)(lVal + K_TOP);       // K_TOP

  const int tid = threadIdx.x;
  const size_t base = (size_t)blockIdx.x * D_DICT;

  for (int j = tid; j < D_DICT; j += TK_NT) rowb[j] = Z[base + j];
  __syncthreads();

  for (int it = 0; it < K_TOP; ++it) {
    float bv = -INFINITY; int bi = 0x7fffffff;
    for (int j = tid; j < D_DICT; j += TK_NT) {
      float v = rowb[j];
      if (v > bv) { bv = v; bi = j; }
    }
    redV[tid] = bv; redI[tid] = bi;
    __syncthreads();
    for (int s = TK_NT / 2; s > 0; s >>= 1) {
      if (tid < s) {
        float ov = redV[tid + s]; int oi = redI[tid + s];
        if (ov > redV[tid] || (ov == redV[tid] && oi < redI[tid])) {
          redV[tid] = ov; redI[tid] = oi;
        }
      }
      __syncthreads();
    }
    if (tid == 0) {
      int bi0 = redI[0]; float bv0 = redV[0];
      lIdx[it] = bi0;
      lVal[it] = bv0 > 0.f ? bv0 : 0.f;   // relu
      rowb[bi0] = -INFINITY;              // marker
    }
    __syncthreads();
  }

  if (tid < K_TOP) {
    outIdx[blockIdx.x * K_TOP + tid] = lIdx[tid];
    outVal[blockIdx.x * K_TOP + tid] = lVal[tid];
  }

  // race-free writeback: each element written exactly once by its owner
  for (int j = tid; j < D_DICT; j += TK_NT) {
    float v = rowb[j];
    float o = 0.f;
    if (v == -INFINITY) {
      for (int t = 0; t < K_TOP; ++t)
        if (lIdx[t] == j) { o = lVal[t]; break; }
    }
    Z[base + j] = o;
  }
}

// ---------------- Sparse decode: x_hat = sparse @ W_dec^T + b_dec -----------
__global__ void sae_decode_kernel(const float* __restrict__ WdT,
                                  const float* __restrict__ b_dec,
                                  const int* __restrict__ idx,
                                  const float* __restrict__ val,
                                  float* __restrict__ Xhat) {
  __shared__ int   sIdx[K_TOP];
  __shared__ float sVal[K_TOP];
  const int n = blockIdx.x, tid = threadIdx.x;   // 256 threads
  if (tid < K_TOP) {
    sIdx[tid] = idx[n * K_TOP + tid];
    sVal[tid] = val[n * K_TOP + tid];
  }
  __syncthreads();
  float acc[4];
  for (int j = 0; j < 4; ++j) acc[j] = b_dec[tid + j * 256];
  for (int t = 0; t < K_TOP; ++t) {
    float v = sVal[t];
    const float* wr = WdT + (size_t)sIdx[t] * D_MODEL;
    for (int j = 0; j < 4; ++j) acc[j] += v * wr[tid + j * 256];
  }
  for (int j = 0; j < 4; ++j)
    Xhat[(size_t)n * D_MODEL + tid + j * 256] = acc[j];
}

// ---------------------------------------------------------------------------
extern "C" void kernel_launch(void* const* d_in, const int* in_sizes, int n_in,
                              void* d_out, int out_size, void* d_ws, size_t ws_size,
                              hipStream_t stream) {
  (void)in_sizes; (void)n_in; (void)out_size; (void)ws_size;
  const float* x     = (const float*)d_in[0];
  const float* W_enc = (const float*)d_in[1];
  const float* b_enc = (const float*)d_in[2];
  const float* W_dec = (const float*)d_in[3];
  const float* b_dec = (const float*)d_in[4];
  // d_in[5] is k (device scalar) == 32, baked in as K_TOP.

  float* xhat   = (float*)d_out;
  float* sparse = (float*)d_out + (size_t)N_TOK * D_MODEL;

  // workspace layout (132 MB total)
  char* ws = (char*)d_ws;
  ushort_t* Xb  = (ushort_t*)(ws);                                   // 32 MB
  ushort_t* Wb  = (ushort_t*)(ws + (size_t)32 * 1024 * 1024);        // 32 MB
  float*    WdT = (float*)   (ws + (size_t)64 * 1024 * 1024);        // 64 MB
  int*      tIdx= (int*)     (ws + (size_t)128 * 1024 * 1024);       //  2 MB
  float*    tVal= (float*)   (ws + (size_t)130 * 1024 * 1024);       //  2 MB

  // 1) fp32 -> bf16 (both operands then live in L2: 32 MB + 32 MB < 192 MB)
  {
    long long nx = (long long)N_TOK * D_MODEL;
    cvt_bf16_kernel<<<(uint32_t)(nx / 4 / 256), 256, 0, stream>>>(x, Xb, nx);
    long long nw = (long long)D_DICT * D_MODEL;
    cvt_bf16_kernel<<<(uint32_t)(nw / 4 / 256), 256, 0, stream>>>(W_enc, Wb, nw);
  }

  // 2) transpose W_dec once so decode gathers are coalesced (64 MB, L2-resident)
  transpose_wdec_kernel<<<dim3(D_DICT / 32, D_MODEL / 32), 256, 0, stream>>>(W_dec, WdT);

  // 3) encode GEMM (bf16 WMMA, fp32 accumulate) -> z written into sparse region
  sae_encode_wmma<<<dim3(D_DICT / BN, N_TOK / BM), 256, 0, stream>>>(Xb, Wb, b_enc, sparse);

  // 4) per-row top-32 + relu, rewrite sparse in place, emit compact lists
  size_t tk_bytes = (size_t)(D_DICT + TK_NT + TK_NT + K_TOP + K_TOP) * 4;
  sae_topk_kernel<<<N_TOK, TK_NT, tk_bytes, stream>>>(sparse, tIdx, tVal);

  // 5) sparse decode -> x_hat
  sae_decode_kernel<<<N_TOK, 256, 0, stream>>>(WdT, b_dec, tIdx, tVal, xhat);
}
